// DfmSW_MSA_47777216200751
// MI455X (gfx1250) — matrix operand order repaired and checked by reading, not compiled, not run on hardware
//
#include <hip/hip_runtime.h>
#include <hip/hip_fp16.h>
#include <math.h>

typedef __attribute__((ext_vector_type(16))) _Float16 v16h;
typedef __attribute__((ext_vector_type(8)))  _Float16 v8h;
typedef __attribute__((ext_vector_type(8)))  float    v8f;

#define H_     128
#define W_     128
#define HP_    130          // halo-padded
#define WP_    130
#define HW_    (H_*W_)
#define C_IN   64
#define INNER  512
#define BATCH  2
#define T_     5
#define HEADS  8
#define DIMH   64
#define SCALE_ 0.125f
#define EPS_   1e-5f

// ---------------------------------------------------------------------------
// 0) Zero the 1-pixel halo ring of a padded NHWC fp16 buffer.
//    516 ring pixels per image (2*130 + 2*128).
// ---------------------------------------------------------------------------
__global__ void zero_halo_kernel(_Float16* __restrict__ buf, int imgs, int C)
{
    int idx = blockIdx.x * blockDim.x + threadIdx.x;
    int per = 516 * C;
    if (idx >= imgs * per) return;
    int img = idx / per;
    int r   = idx - img * per;
    int h   = r / C;
    int c   = r - h * C;
    int y, x;
    if (h < WP_)          { y = 0;       x = h; }
    else if (h < 2 * WP_) { y = HP_ - 1; x = h - WP_; }
    else { int q = h - 2 * WP_; y = 1 + (q >> 1); x = (q & 1) ? (WP_ - 1) : 0; }
    buf[(((size_t)img * HP_ + y) * WP_ + x) * C + c] = (_Float16)0.f;
}

// ---------------------------------------------------------------------------
// 1) Bilinear warp (grid_sample, zeros) + LayerNorm(C) of kv frames.
//    Input NCHW f32; outputs PADDED NHWC fp16 (interior only).
// ---------------------------------------------------------------------------
__global__ void warp_ln_kv_kernel(const float* __restrict__ kv,    // [B*T,64,H,W]
                                  const float* __restrict__ kpe,   // [B*T,1,H,W]
                                  const float* __restrict__ flow,  // [B*T,2,H,W]
                                  const float* __restrict__ g,
                                  const float* __restrict__ bta,
                                  _Float16* __restrict__ kvn,      // [B*T,HP,WP,64]
                                  _Float16* __restrict__ kpen)     // [B*T,HP,WP,64]
{
    int pix = blockIdx.x * blockDim.x + threadIdx.x;
    if (pix >= BATCH * T_ * HW_) return;
    int img = pix / HW_;
    int p   = pix - img * HW_;
    int y   = p / W_;
    int x   = p - y * W_;

    const float* fb = flow + (size_t)img * 2 * HW_;
    float fx = fb[p];
    float fy = fb[HW_ + p];
    // reference: ix = vx*W/(W-1) - 0.5 (align_corners normalize + default unnormalize)
    float ix = ((float)x + fx) * ((float)W_ / (float)(W_ - 1)) - 0.5f;
    float iy = ((float)y + fy) * ((float)H_ / (float)(H_ - 1)) - 0.5f;
    float x0f = floorf(ix), y0f = floorf(iy);
    int   x0 = (int)x0f,    y0 = (int)y0f;
    float ax = ix - x0f,    ay = iy - y0f;
    float w00 = (1.f - ax) * (1.f - ay);
    float w10 = ax * (1.f - ay);
    float w01 = (1.f - ax) * ay;
    float w11 = ax * ay;
    bool inx0 = (x0 >= 0) && (x0 <= W_ - 1);
    bool inx1 = (x0 + 1 >= 0) && (x0 + 1 <= W_ - 1);
    bool iny0 = (y0 >= 0) && (y0 <= H_ - 1);
    bool iny1 = (y0 + 1 >= 0) && (y0 + 1 <= H_ - 1);
    float m00 = (inx0 && iny0) ? w00 : 0.f;
    float m10 = (inx1 && iny0) ? w10 : 0.f;
    float m01 = (inx0 && iny1) ? w01 : 0.f;
    float m11 = (inx1 && iny1) ? w11 : 0.f;
    int xc0 = min(max(x0, 0), W_ - 1),     xc1 = min(max(x0 + 1, 0), W_ - 1);
    int yc0 = min(max(y0, 0), H_ - 1),     yc1 = min(max(y0 + 1, 0), H_ - 1);
    int i00 = yc0 * W_ + xc0, i10 = yc0 * W_ + xc1;
    int i01 = yc1 * W_ + xc0, i11 = yc1 * W_ + xc1;

    const float* src = kv + (size_t)img * C_IN * HW_;
    float vals[C_IN];
    float s1 = 0.f, s2 = 0.f;
#pragma unroll
    for (int c = 0; c < C_IN; ++c) {
        const float* pc = src + (size_t)c * HW_;
        float v = m00 * pc[i00] + m10 * pc[i10] + m01 * pc[i01] + m11 * pc[i11];
        vals[c] = v;
        s1 += v;
        s2 += v * v;
    }
    float mu  = s1 * (1.f / C_IN);
    float var = s2 * (1.f / C_IN) - mu * mu;
    float inv = rsqrtf(var + EPS_);
    float pe  = kpe[(size_t)img * HW_ + p];
    size_t ob = (((size_t)img * HP_ + y + 1) * WP_ + x + 1) * C_IN;   // padded NHWC
#pragma unroll
    for (int c = 0; c < C_IN; ++c) {
        float nv = (vals[c] - mu) * inv * g[c] + bta[c];
        kvn [ob + c] = (_Float16)nv;
        kpen[ob + c] = (_Float16)(nv + pe);
    }
}

// ---------------------------------------------------------------------------
// 2) LayerNorm(C) of q + q_pe broadcast -> PADDED NHWC fp16.
// ---------------------------------------------------------------------------
__global__ void ln_q_kernel(const float* __restrict__ q,    // [B,64,H,W]
                            const float* __restrict__ qpe,  // [B,1,H,W]
                            const float* __restrict__ g,
                            const float* __restrict__ bta,
                            _Float16* __restrict__ qpen)    // [B,HP,WP,64]
{
    int pix = blockIdx.x * blockDim.x + threadIdx.x;
    if (pix >= BATCH * HW_) return;
    int img = pix / HW_;
    int p   = pix - img * HW_;
    int y   = p / W_;
    int x   = p - y * W_;
    const float* src = q + (size_t)img * C_IN * HW_;
    float vals[C_IN];
    float s1 = 0.f, s2 = 0.f;
#pragma unroll
    for (int c = 0; c < C_IN; ++c) {
        float v = src[(size_t)c * HW_ + p];
        vals[c] = v;
        s1 += v;
        s2 += v * v;
    }
    float mu  = s1 * (1.f / C_IN);
    float var = s2 * (1.f / C_IN) - mu * mu;
    float inv = rsqrtf(var + EPS_);
    float pe  = qpe[(size_t)img * HW_ + p];
    size_t ob = (((size_t)img * HP_ + y + 1) * WP_ + x + 1) * C_IN;
#pragma unroll
    for (int c = 0; c < C_IN; ++c) {
        float nv = (vals[c] - mu) * inv * g[c] + bta[c];
        qpen[ob + c] = (_Float16)(nv + pe);
    }
}

// ---------------------------------------------------------------------------
// 3) Repack conv weights [OC,IC,3,3] f32 -> fp16 CDNA5 A-matrix tiles with
//    TAP-MAJOR K ordering: k = tap*IC + ic (tap = ty*3+tx).
//    A[m][K] with K(e) = kb*32 + (lane>>4)*16 + e, m = lane&15.
// ---------------------------------------------------------------------------
__global__ void pack_w_kernel(const float* __restrict__ w, _Float16* __restrict__ wp,
                              int IC, int OC, int KB)
{
    int idx = blockIdx.x * blockDim.x + threadIdx.x;
    int total = (OC / 16) * KB * 32 * 16;
    if (idx >= total) return;
    int e    = idx & 15;
    int lane = (idx >> 4) & 31;
    int kb   = (idx >> 9) % KB;
    int ot   = idx / (KB * 512);
    int m     = lane & 15;
    int khalf = lane >> 4;
    int k  = kb * 32 + khalf * 16 + e;   // tap-major k
    int oc = ot * 16 + m;
    int tap = k / IC;
    int ic  = k - tap * IC;
    float v = (tap < 9) ? w[((size_t)oc * IC + ic) * 9 + tap] : 0.f;
    wp[idx] = (_Float16)v;
}

// ---------------------------------------------------------------------------
// 4) Implicit-GEMM 3x3 conv, halo-padded NHWC fp16 in -> no bounds checks.
//    Wave tile: 16 oc x 64 px (4 B-frags reuse 1 A-frag -> 4 WMMA / K-step).
//    Software-pipelined: next K-step's A + B fragments are prefetched into a
//    second register set while the current WMMA group executes, so VMEM
//    latency overlaps the matrix pipe instead of serializing before it.
// ---------------------------------------------------------------------------
template <int IC, int OC, int WRITE_F32>
__global__ __launch_bounds__(128)
void conv3x3_wmma_kernel(const _Float16* __restrict__ in,  // [img,HP,WP,IC] padded
                         const _Float16* __restrict__ wp,  // packed A tiles
                         _Float16* __restrict__ out16,     // [img,H,W,OC]
                         float* __restrict__ out32)        // [img,OC,H,W]
{
    constexpr int KB = (IC * 9) / 32;
    const int tile = blockIdx.x;                  // H * (W/64) tiles per image
    const int img  = blockIdx.z;
    const int wave = threadIdx.x >> 5;
    const int lane = threadIdx.x & 31;
    const int oc_tile = blockIdx.y * 4 + wave;
    const int y  = tile >> 1;
    const int xb = (tile & 1) << 6;               // 0 or 64
    const int n     = lane & 15;
    const int khalf = lane >> 4;

    const _Float16* abase = wp + (size_t)oc_tile * KB * 512 + (size_t)lane * 16;
    // padded coords: conv tap (ty,tx) reads (y+ty, xb+n+tx) in padded space
    const _Float16* ibase = in + (((size_t)img * HP_ + y) * WP_ + xb + n) * IC + khalf * 16;

    auto brow = [&](int kb) -> const _Float16* {
        const int tap = (kb * 32) / IC;           // constant-folded shift
        const int icb = (kb * 32) - tap * IC;
        const int ty  = tap / 3, tx = tap - ty * 3;
        return ibase + ((size_t)ty * WP_ + tx) * IC + icb;
    };

    // prologue: stage K-step 0
    const _Float16* r0 = brow(0);
    v16h a  = *(const v16h*)(abase);
    v16h b0 = *(const v16h*)(r0);
    v16h b1 = *(const v16h*)(r0 + (size_t)16 * IC);
    v16h b2 = *(const v16h*)(r0 + (size_t)32 * IC);
    v16h b3 = *(const v16h*)(r0 + (size_t)48 * IC);

    v8f acc0 = {}, acc1 = {}, acc2 = {}, acc3 = {};
#pragma unroll 2
    for (int kb = 0; kb < KB - 1; ++kb) {
        // prefetch K-step kb+1 into the shadow set (independent of cur)
        const _Float16* rn = brow(kb + 1);
        v16h an  = *(const v16h*)(abase + (size_t)(kb + 1) * 512);
        v16h bn0 = *(const v16h*)(rn);
        v16h bn1 = *(const v16h*)(rn + (size_t)16 * IC);
        v16h bn2 = *(const v16h*)(rn + (size_t)32 * IC);
        v16h bn3 = *(const v16h*)(rn + (size_t)48 * IC);

        acc0 = __builtin_amdgcn_wmma_f32_16x16x32_f16(false, a, false, b0, (short)0, acc0, false, false);
        acc1 = __builtin_amdgcn_wmma_f32_16x16x32_f16(false, a, false, b1, (short)0, acc1, false, false);
        acc2 = __builtin_amdgcn_wmma_f32_16x16x32_f16(false, a, false, b2, (short)0, acc2, false, false);
        acc3 = __builtin_amdgcn_wmma_f32_16x16x32_f16(false, a, false, b3, (short)0, acc3, false, false);

        a = an; b0 = bn0; b1 = bn1; b2 = bn2; b3 = bn3;
    }
    // epilogue: last K-step
    acc0 = __builtin_amdgcn_wmma_f32_16x16x32_f16(false, a, false, b0, (short)0, acc0, false, false);
    acc1 = __builtin_amdgcn_wmma_f32_16x16x32_f16(false, a, false, b1, (short)0, acc1, false, false);
    acc2 = __builtin_amdgcn_wmma_f32_16x16x32_f16(false, a, false, b2, (short)0, acc2, false, false);
    acc3 = __builtin_amdgcn_wmma_f32_16x16x32_f16(false, a, false, b3, (short)0, acc3, false, false);

    // D layout: VGPR r holds M = r + 8*khalf; N = lane&15 -> oc block of 8
    const int ocb = oc_tile * 16 + khalf * 8;
    v8f accs[4] = {acc0, acc1, acc2, acc3};
#pragma unroll
    for (int t = 0; t < 4; ++t) {
        if (WRITE_F32) {
            size_t pp = (size_t)y * W_ + xb + t * 16 + n;
#pragma unroll
            for (int r = 0; r < 8; ++r)
                out32[((size_t)img * OC + ocb + r) * HW_ + pp] = accs[t][r];
        } else {
            size_t pixIdx = ((size_t)img * H_ + y) * W_ + xb + t * 16 + n;
            v8h h;
#pragma unroll
            for (int r = 0; r < 8; ++r) h[r] = (_Float16)accs[t][r];
            *(v8h*)(out16 + pixIdx * OC + ocb) = h;   // 16B packed NHWC store
        }
    }
}

// ---------------------------------------------------------------------------
// 5) 2x2 deformable-window attention, NHWC fp16 activations.
//    One wave per (window, head): lanes 0..19 own a kv token for sim (contig
//    channel reads), serial softmax over 20, then lane = dim for attn @ V.
//    Output 'of' is halo-padded (input of the final conv).
// ---------------------------------------------------------------------------
__global__ void win_attn_kernel(const _Float16* __restrict__ qf,  // [B,H,W,512]
                                const _Float16* __restrict__ kf,  // [B*T,H,W,512]
                                const _Float16* __restrict__ vf,  // [B*T,H,W,512]
                                _Float16* __restrict__ of)        // [B,HP,WP,512]
{
    __shared__ float qs [HEADS][4][DIMH];
    __shared__ float att[HEADS][4][20];
    const int win  = blockIdx.x;
    const int head = threadIdx.x >> 5;
    const int lane = threadIdx.x & 31;
    const int bidx = win >> 12;
    const int rem  = win & 4095;
    const int wy = rem >> 6, wx = rem & 63;

    // stage q * SCALE into LDS (8 values per lane), NHWC reads
#pragma unroll
    for (int e = 0; e < 8; ++e) {
        int idx = lane * 8 + e;          // 0..255
        int qi  = idx >> 6;
        int d   = idx & 63;
        int py = wy * 2 + (qi >> 1), px = wx * 2 + (qi & 1);
        float qv = (float)qf[(((size_t)bidx * H_ + py) * W_ + px) * INNER + head * DIMH + d];
        qs[head][qi][d] = qv * SCALE_;
    }
    __syncthreads();

    // sim[qi][kj] on lanes 0..19 (contiguous 64-channel run per token)
    if (lane < 20) {
        int f = lane >> 2, pq = lane & 3;
        int py = wy * 2 + (pq >> 1), px = wx * 2 + (pq & 1);
        int imgk = bidx * T_ + f;
        const _Float16* kp = kf + (((size_t)imgk * H_ + py) * W_ + px) * INNER + head * DIMH;
        float s0 = 0.f, s1 = 0.f, s2 = 0.f, s3 = 0.f;
#pragma unroll 8
        for (int d = 0; d < DIMH; ++d) {
            float kv = (float)kp[d];
            s0 += qs[head][0][d] * kv;
            s1 += qs[head][1][d] * kv;
            s2 += qs[head][2][d] * kv;
            s3 += qs[head][3][d] * kv;
        }
        att[head][0][lane] = s0;
        att[head][1][lane] = s1;
        att[head][2][lane] = s2;
        att[head][3][lane] = s3;
    }
    __syncthreads();

    // softmax over 20 kv tokens; lanes 0..3 each handle one q token
    if (lane < 4) {
        float mx = -1e30f;
        for (int j = 0; j < 20; ++j) mx = fmaxf(mx, att[head][lane][j]);
        float sum = 0.f;
        for (int j = 0; j < 20; ++j) {
            float e = __expf(att[head][lane][j] - mx);
            att[head][lane][j] = e;
            sum += e;
        }
        float r = 1.f / sum;
        for (int j = 0; j < 20; ++j) att[head][lane][j] *= r;
    }
    __syncthreads();

    // o = attn @ V; lane owns dims {lane, lane+32}; coalesced NHWC reads
    float o[4][2] = {};
    for (int kj = 0; kj < 20; ++kj) {
        int f = kj >> 2, pq = kj & 3;
        int py = wy * 2 + (pq >> 1), px = wx * 2 + (pq & 1);
        int imgk = bidx * T_ + f;
        const _Float16* vp = vf + (((size_t)imgk * H_ + py) * W_ + px) * INNER + head * DIMH;
        float v0 = (float)vp[lane];
        float v1 = (float)vp[lane + 32];
#pragma unroll
        for (int qi = 0; qi < 4; ++qi) {
            float a = att[head][qi][kj];
            o[qi][0] += a * v0;
            o[qi][1] += a * v1;
        }
    }
#pragma unroll
    for (int qi = 0; qi < 4; ++qi) {
        int py = wy * 2 + (qi >> 1), px = wx * 2 + (qi & 1);
        _Float16* op = of + (((size_t)bidx * HP_ + py + 1) * WP_ + px + 1) * INNER + head * DIMH;
        op[lane]      = (_Float16)o[qi][0];
        op[lane + 32] = (_Float16)o[qi][1];
    }
}

// ---------------------------------------------------------------------------
// Launch
// ---------------------------------------------------------------------------
extern "C" void kernel_launch(void* const* d_in, const int* in_sizes, int n_in,
                              void* d_out, int out_size, void* d_ws, size_t ws_size,
                              hipStream_t stream)
{
    const float* q_inp  = (const float*)d_in[0];
    const float* q_pe   = (const float*)d_in[1];
    const float* kv_inp = (const float*)d_in[2];
    const float* k_pe   = (const float*)d_in[3];
    const float* k_flow = (const float*)d_in[4];
    const float* wq     = (const float*)d_in[5];
    const float* wk     = (const float*)d_in[6];
    const float* wv     = (const float*)d_in[7];
    const float* wo     = (const float*)d_in[8];
    const float* gq     = (const float*)d_in[9];
    const float* bq     = (const float*)d_in[10];
    const float* gkv    = (const float*)d_in[11];
    const float* bkv    = (const float*)d_in[12];
    float* out = (float*)d_out;

    // workspace layout (fp16 elems; all sizes multiples of 16 -> 32B aligned)
    _Float16* wsh = (_Float16*)d_ws;
    const size_t PPIX   = (size_t)HP_ * WP_;                 // padded pixels
    const size_t QPEN_E = (size_t)BATCH * PPIX * C_IN;       // padded
    const size_t KVN_E  = (size_t)BATCH * T_ * PPIX * C_IN;  // padded
    const size_t OF_E   = (size_t)BATCH * PPIX * INNER;      // padded
    const size_t QF_E   = (size_t)BATCH * HW_ * INNER;
    const size_t KF_E   = (size_t)BATCH * T_ * HW_ * INNER;
    const size_t WPK_E  = 32u * 18u * 512u;
    const size_t WPO_E  = 4u * 144u * 512u;

    _Float16* qpen = wsh;              size_t off = QPEN_E;
    _Float16* kvn  = wsh + off;        off += KVN_E;
    _Float16* kpen = wsh + off;        off += KVN_E;
    _Float16* of   = wsh + off;        off += OF_E;
    _Float16* qf   = wsh + off;        off += QF_E;
    _Float16* kf   = wsh + off;        off += KF_E;
    _Float16* vf   = wsh + off;        off += KF_E;
    _Float16* wqp  = wsh + off;        off += WPK_E;
    _Float16* wkp  = wsh + off;        off += WPK_E;
    _Float16* wvp  = wsh + off;        off += WPK_E;
    _Float16* wop  = wsh + off;        off += WPO_E;

    // 0) zero halo rings of all padded conv inputs
    {
        int n1 = BATCH * 516 * C_IN;
        zero_halo_kernel<<<(n1 + 255) / 256, 256, 0, stream>>>(qpen, BATCH, C_IN);
        int n2 = BATCH * T_ * 516 * C_IN;
        zero_halo_kernel<<<(n2 + 255) / 256, 256, 0, stream>>>(kvn,  BATCH * T_, C_IN);
        zero_halo_kernel<<<(n2 + 255) / 256, 256, 0, stream>>>(kpen, BATCH * T_, C_IN);
        int n3 = BATCH * 516 * INNER;
        zero_halo_kernel<<<(n3 + 255) / 256, 256, 0, stream>>>(of,   BATCH, INNER);
    }

    // 1) normalize inputs (NCHW f32 -> padded NHWC f16)
    ln_q_kernel<<<(BATCH * HW_ + 255) / 256, 256, 0, stream>>>(q_inp, q_pe, gq, bq, qpen);
    warp_ln_kv_kernel<<<(BATCH * T_ * HW_ + 255) / 256, 256, 0, stream>>>(
        kv_inp, k_pe, k_flow, gkv, bkv, kvn, kpen);

    // 2) pack weights into WMMA A-tile layout (tap-major K)
    const int nk = (int)WPK_E, no = (int)WPO_E;
    pack_w_kernel<<<(nk + 255) / 256, 256, 0, stream>>>(wq, wqp, C_IN, INNER, 18);
    pack_w_kernel<<<(nk + 255) / 256, 256, 0, stream>>>(wk, wkp, C_IN, INNER, 18);
    pack_w_kernel<<<(nk + 255) / 256, 256, 0, stream>>>(wv, wvp, C_IN, INNER, 18);
    pack_w_kernel<<<(no + 255) / 256, 256, 0, stream>>>(wo, wop, INNER, C_IN, 144);

    // 3) QKV convs: grid = (H * W/64 tiles, OC/64, images), 4 WMMA / K-step
    dim3 blk(128, 1, 1);
    conv3x3_wmma_kernel<C_IN, INNER, 0><<<dim3(256, INNER / 64, BATCH), blk, 0, stream>>>(
        qpen, wqp, qf, nullptr);
    conv3x3_wmma_kernel<C_IN, INNER, 0><<<dim3(256, INNER / 64, BATCH * T_), blk, 0, stream>>>(
        kpen, wkp, kf, nullptr);
    conv3x3_wmma_kernel<C_IN, INNER, 0><<<dim3(256, INNER / 64, BATCH * T_), blk, 0, stream>>>(
        kvn, wvp, vf, nullptr);

    // 4) windowed attention: 8192 windows x 8 heads (wave per head)
    win_attn_kernel<<<dim3(BATCH * 64 * 64), dim3(256), 0, stream>>>(qf, kf, vf, of);

    // 5) output conv 512 -> 64, f32 NCHW straight into d_out
    conv3x3_wmma_kernel<INNER, C_IN, 1><<<dim3(256, 1, BATCH), blk, 0, stream>>>(
        of, wop, nullptr, out);
}
